// RN_58437325029705
// MI455X (gfx1250) — compile-verified, matrix operand order checked by reading, not run on hardware
//
#include <hip/hip_runtime.h>
#include <stdint.h>
#include <math.h>

// ---------------------------------------------------------------------------
// MI455X (gfx1250) fused implementation.
//
// Roofline reasoning:
//  * box_feats = 64*512*2048 f32 = 256 MB -> ~11us @ 23.3 TB/s if read ONCE.
//    => fuse both projections (W and Wp) + activations + fw-dot + masked sum
//       into a single pass over box_feats.
//  * 137 GFLOP of GEMM -> bf16 WMMA (v_wmma_f32_16x16x32_bf16) with f32
//    accumulation; weights pre-swizzled to B-fragment layout (4 MB bf16,
//    L2-resident; per-lane 32B contiguous loads).
//  * LSTM: x@Wih^T hoisted out of the time loop (one [896,320]x[320,4096]
//    WMMA GEMM on zero-padded K); only h@Whh^T stays sequential.
//  * All GEMM A-staging is branch-free (K pre-padded to a multiple of 32),
//    and f32->bf16 conversion uses packed hardware converts.
// ---------------------------------------------------------------------------

typedef __attribute__((ext_vector_type(16))) __bf16 v16bf;
typedef __attribute__((ext_vector_type(2)))  __bf16 bf16x2;
typedef __attribute__((ext_vector_type(8)))  float  v8f;
typedef __attribute__((ext_vector_type(2)))  float  f32x2;

__device__ __forceinline__ v8f wmma_bf16(v16bf a, v16bf b, v8f c) {
  return __builtin_amdgcn_wmma_f32_16x16x32_bf16(false, a, false, b,
                                                 (short)0, c, false, false);
}

// packed f32 -> bf16 conversion (lets the backend select v_cvt_pk_bf16_f32)
__device__ __forceinline__ uint32_t pack_bf16x2(float lo, float hi) {
  bf16x2 v = __builtin_convertvector((f32x2){lo, hi}, bf16x2);
  return __builtin_bit_cast(uint32_t, v);
}
__device__ __forceinline__ unsigned short f32_to_bf16(float f) {
  __bf16 v = (__bf16)f;
  return __builtin_bit_cast(unsigned short, v);
}
__device__ __forceinline__ float sigf(float x) {
  return 1.0f / (1.0f + __expf(-x));
}

// --------------------------- tiny utility kernels ---------------------------

__global__ __launch_bounds__(256) void k_zero(float* p, int n) {
  int i = blockIdx.x * 256 + threadIdx.x;
  if (i < n) p[i] = 0.0f;
}

// zero-padded copy: dst[r*ldd + k] = (k < Ksrc) ? src[r*Ksrc + k] : 0
__global__ __launch_bounds__(256) void k_padcopy(const float* __restrict__ src,
                                                 int Ksrc, float* __restrict__ dst,
                                                 int ldd, long total) {
  long id = (long)blockIdx.x * 256 + threadIdx.x;
  if (id >= total) return;
  long r = id / ldd;
  int k = (int)(id - r * ldd);
  dst[id] = (k < Ksrc) ? src[r * (long)Ksrc + k] : 0.0f;
}

// per-row L2 norm scale: S[row] = G[row] / ||V[row,:]||
__global__ __launch_bounds__(256) void k_rownorm(const float* __restrict__ V,
                                                 const float* __restrict__ G,
                                                 float* __restrict__ S, int ncols) {
  __shared__ float red[256];
  int row = blockIdx.x;
  const float* v = V + (size_t)row * ncols;
  float s = 0.0f;
  for (int k = threadIdx.x; k < ncols; k += 256) { float x = v[k]; s += x * x; }
  red[threadIdx.x] = s; __syncthreads();
  for (int o = 128; o > 0; o >>= 1) {
    if (threadIdx.x < o) red[threadIdx.x] += red[threadIdx.x + o];
    __syncthreads();
  }
  if (threadIdx.x == 0) S[row] = G[row] * rsqrtf(red[0]);
}

// fw = f_g * f_v / ||f_v||  (512 elems, one block)
__global__ __launch_bounds__(256) void k_fwnorm(const float* __restrict__ fv,
                                                const float* __restrict__ fg,
                                                float* __restrict__ fwn) {
  __shared__ float red[256];
  __shared__ float sc;
  float a = fv[threadIdx.x], b = fv[threadIdx.x + 256];
  red[threadIdx.x] = a * a + b * b; __syncthreads();
  for (int o = 128; o > 0; o >>= 1) {
    if (threadIdx.x < o) red[threadIdx.x] += red[threadIdx.x + o];
    __syncthreads();
  }
  if (threadIdx.x == 0) sc = fg[0] * rsqrtf(red[0]);
  __syncthreads();
  fwn[threadIdx.x]       = a * sc;
  fwn[threadIdx.x + 256] = b * sc;
}

// ------------- bf16 B-fragment swizzle (WMMA 16x16x32 layout) ---------------
// dst[((ks*CT + ct)*32 + lane)*16 + e] with
//   col = ct*16 + (lane&15)
//   k   = ks*32 + (e<8 ? e : e+8) + (lane>=16 ? 8 : 0)
__global__ __launch_bounds__(256) void k_swizzle(const float* __restrict__ src,
                                                 int rowstride, int Kvalid,
                                                 const float* __restrict__ scale,
                                                 unsigned short* __restrict__ dst,
                                                 int CT, int Kpad) {
  long id = (long)blockIdx.x * 256 + threadIdx.x;
  long total = (long)CT * 16 * Kpad;
  if (id >= total) return;
  int e = (int)(id & 15);
  int lane = (int)((id >> 4) & 31);
  long q = id >> 9;
  int ct = (int)(q % CT);
  int ks = (int)(q / CT);
  int col = ct * 16 + (lane & 15);
  int k = ks * 32 + ((e < 8) ? e : e + 8) + ((lane >= 16) ? 8 : 0);
  float v = 0.0f;
  if (k < Kvalid) {
    v = src[(size_t)col * rowstride + k];
    if (scale) v *= scale[col];
  }
  dst[id] = f32_to_bf16(v);
}

// ------------------------ generic bf16 WMMA GEMM ----------------------------
// out[M,N] = A[M,Kpad](f32, K pre-padded, converted on the fly) x Bswiz^T
// (+ addends). grid = (M/16, N/128); 8 waves, one 16x16 tile each.
__global__ __launch_bounds__(256) void k_gemm(const float* __restrict__ A, int lda,
                                              int Kpad,
                                              const unsigned short* __restrict__ Bw,
                                              int CT,
                                              const float* __restrict__ add0,
                                              long add0stride,
                                              const float* __restrict__ add1,
                                              float* __restrict__ out, int ldo) {
  __shared__ alignas(32) unsigned short ldsA[32 * 16];  // one 16x32 bf16 A frag set
  int tid = threadIdx.x, wave = tid >> 5, lane = tid & 31;
  int rt0 = blockIdx.x * 16;
  int ct = blockIdx.y * 8 + wave;
  v8f acc = {0.f, 0.f, 0.f, 0.f, 0.f, 0.f, 0.f, 0.f};

  int sl = (tid >> 1) & 31, half = tid & 1;
  bool stager = tid < 64;
  const float* Arow = A + (size_t)(rt0 + (sl & 15)) * lda;
  int kb = half * 16 + ((sl >= 16) ? 8 : 0);
  unsigned short* sdst = &ldsA[sl * 16 + half * 8];

  int ksteps = Kpad >> 5;
  for (int ks = 0; ks < ksteps; ++ks) {
    __syncthreads();
    if (stager) {
      const float* sp = Arow + ks * 32 + kb;
      float4 f0 = *(const float4*)sp;
      float4 f1 = *(const float4*)(sp + 4);
      uint4 u;
      u.x = pack_bf16x2(f0.x, f0.y); u.y = pack_bf16x2(f0.z, f0.w);
      u.z = pack_bf16x2(f1.x, f1.y); u.w = pack_bf16x2(f1.z, f1.w);
      *(uint4*)sdst = u;
    }
    __syncthreads();
    v16bf a = *((const v16bf*)ldsA + lane);
    v16bf b = *((const v16bf*)Bw + ((size_t)ks * CT + ct) * 32 + lane);
    acc = wmma_bf16(a, b, acc);
  }

  int col = ct * 16 + (lane & 15);
  int rhi = (lane >> 4) * 8;
#pragma unroll
  for (int v = 0; v < 8; ++v) {
    int grow = rt0 + rhi + v;
    float addv = 0.0f;
    if (add0) addv += add0[(size_t)grow * add0stride + col];
    if (add1) addv += add1[col];
    out[(size_t)grow * ldo + col] = acc[v] + addv;
  }
}

// ------------------------------- LSTM cell ----------------------------------
__global__ __launch_bounds__(256) void k_cell(const float* __restrict__ gates,
                                              float* __restrict__ h,
                                              float* __restrict__ c) {
  int id = blockIdx.x * 256 + threadIdx.x;  // 64*1024
  int b = id >> 10, j = id & 1023;
  size_t base = (size_t)b * 4096;
  float gi = gates[base + j];
  float gf = gates[base + 1024 + j];
  float gg = gates[base + 2048 + j];
  float go = gates[base + 3072 + j];
  float cc = sigf(gf) * c[id] + sigf(gi) * tanhf(gg);
  c[id] = cc;
  h[id] = sigf(go) * tanhf(cc);
}

// -------- q projection: qy/qg[b,h] = (q_rnn[b,:] . Wv[h,2048:]) * s + bias --
__global__ __launch_bounds__(256) void k_qproj(const float* __restrict__ h,
                                               const float* __restrict__ Wv,
                                               const float* __restrict__ Wpv,
                                               const float* __restrict__ sW,
                                               const float* __restrict__ sWp,
                                               const float* __restrict__ Wb,
                                               const float* __restrict__ Wpb,
                                               float* __restrict__ qy,
                                               float* __restrict__ qg) {
  int id = blockIdx.x * 256 + threadIdx.x;  // 2*64*512
  int hh = id & 511, b = (id >> 9) & 63, m = id >> 15;
  const float* V = m ? Wpv : Wv;
  const float* S = m ? sWp : sW;
  const float* Bb = m ? Wpb : Wb;
  float* o = m ? qg : qy;
  const float* hr = h + (size_t)b * 1024;
  const float* vr = V + (size_t)hh * 3072 + 2048;
  float d = 0.0f;
  for (int k = 0; k < 1024; ++k) d += hr[k] * vr[k];
  o[(size_t)b * 512 + hh] = d * S[hh] + Bb[hh];
}

// --------------------- fused main pass over box_feats -----------------------
// One workgroup = 64 rows (one batch b) x 512 cols x {y,g}.
// 8 waves, each: 4 row-tiles x 4 col-tiles x 2 matrices = 32 WMMA accum tiles.
__global__ __launch_bounds__(256) void k_fused(const float* __restrict__ box,
                                               const unsigned short* __restrict__ wY,
                                               const unsigned short* __restrict__ wG,
                                               const float* __restrict__ qy,
                                               const float* __restrict__ qg,
                                               const float* __restrict__ fwn,
                                               const int* __restrict__ indexv,
                                               float* __restrict__ counts) {
  __shared__ alignas(32) unsigned short ldsA[4 * 32 * 16];  // 4 row-tile frag sets
  __shared__ float rowdot[64];
  __shared__ float redbuf[64];

  int tid = threadIdx.x, wave = tid >> 5, lane = tid & 31;
  long wgrow0 = (long)blockIdx.x * 64;
  int bidx = (int)(wgrow0 >> 9);          // 64 rows never straddle a batch
  if (tid < 64) rowdot[tid] = 0.0f;

  v8f acc[4][4][2];
#pragma unroll
  for (int rt = 0; rt < 4; ++rt)
#pragma unroll
    for (int j = 0; j < 4; ++j)
#pragma unroll
      for (int m = 0; m < 2; ++m)
        acc[rt][j][m] = (v8f){0.f, 0.f, 0.f, 0.f, 0.f, 0.f, 0.f, 0.f};

  // staging role: 4 rt x 32 lanes x 2 halves = 256 threads
  int srt = tid >> 6, rem = tid & 63, sl = rem >> 1, shalf = rem & 1;
  const float* srcrow = box + (size_t)(wgrow0 + srt * 16 + (sl & 15)) * 2048;
  int kb = shalf * 16 + ((sl >= 16) ? 8 : 0);
  unsigned short* sdst = &ldsA[(srt * 32 + sl) * 16 + shalf * 8];

  for (int ks = 0; ks < 64; ++ks) {
    __syncthreads();
    {
      const float* sp = srcrow + ks * 32 + kb;
      float4 f0 = *(const float4*)sp;
      float4 f1 = *(const float4*)(sp + 4);
      if (ks < 63) __builtin_prefetch(sp + 32, 0, 1);   // global_prefetch_b8
      uint4 u;
      u.x = pack_bf16x2(f0.x, f0.y); u.y = pack_bf16x2(f0.z, f0.w);
      u.z = pack_bf16x2(f1.x, f1.y); u.w = pack_bf16x2(f1.z, f1.w);
      *(uint4*)sdst = u;
    }
    __syncthreads();

    v16bf bY[4], bG[4];
#pragma unroll
    for (int j = 0; j < 4; ++j) {
      size_t boff = ((size_t)(ks * 32 + (wave * 4 + j)) * 32 + lane);
      bY[j] = *((const v16bf*)wY + boff);
      bG[j] = *((const v16bf*)wG + boff);
    }
#pragma unroll
    for (int rt = 0; rt < 4; ++rt) {
      v16bf a = *((const v16bf*)ldsA + rt * 32 + lane);
#pragma unroll
      for (int j = 0; j < 4; ++j) {
        acc[rt][j][0] = wmma_bf16(a, bY[j], acc[rt][j][0]);
        acc[rt][j][1] = wmma_bf16(a, bG[j], acc[rt][j][1]);
      }
    }
  }

  // epilogue: si = tanh(y)*sigmoid(g); rowdot[r] += si * fw[col]
  int rhi = (lane >> 4) * 8;
  int cl = lane & 15;
  const float* qyb = qy + (size_t)bidx * 512;
  const float* qgb = qg + (size_t)bidx * 512;
#pragma unroll
  for (int rt = 0; rt < 4; ++rt) {
#pragma unroll
    for (int v = 0; v < 8; ++v) {
      int rloc = rt * 16 + rhi + v;
      float s = 0.0f;
#pragma unroll
      for (int j = 0; j < 4; ++j) {
        int col = (wave * 4 + j) * 16 + cl;
        float y = acc[rt][j][0][v] + qyb[col];
        float g = acc[rt][j][1][v] + qgb[col];
        s += tanhf(y) * sigf(g) * fwn[col];
      }
      atomicAdd(&rowdot[rloc], s);   // ds_add_f32
    }
  }
  __syncthreads();

  if (tid < 64) {
    int n = (int)((wgrow0 + tid) & 511);
    float wsi = sigf(rowdot[tid]);
    redbuf[tid] = (n < indexv[bidx]) ? wsi : 0.0f;
  }
  __syncthreads();
  if (tid == 0) {
    float s = 0.0f;
    for (int i = 0; i < 64; ++i) s += redbuf[i];
    atomicAdd(&counts[bidx], s);
  }
}

// ---------------------------------------------------------------------------

extern "C" void kernel_launch(void* const* d_in, const int* in_sizes, int n_in,
                              void* d_out, int out_size, void* d_ws, size_t ws_size,
                              hipStream_t stream) {
  (void)in_sizes; (void)n_in; (void)out_size; (void)ws_size;

  const float* box_feats = (const float*)d_in[2];
  const float* q_feats   = (const float*)d_in[3];
  const int*   indexv    = (const int*)d_in[5];
  const float* W_v  = (const float*)d_in[6];
  const float* W_g  = (const float*)d_in[7];
  const float* W_b  = (const float*)d_in[8];
  const float* Wp_v = (const float*)d_in[9];
  const float* Wp_g = (const float*)d_in[10];
  const float* Wp_b = (const float*)d_in[11];
  const float* f_v  = (const float*)d_in[12];
  const float* f_g  = (const float*)d_in[13];
  const float* Wih  = (const float*)d_in[14];
  const float* Whh  = (const float*)d_in[15];
  const float* bih  = (const float*)d_in[16];
  const float* bhh  = (const float*)d_in[17];
  float* counts = (float*)d_out;

  // workspace carve-out (256B aligned)
  char* base = (char*)d_ws;
  size_t off = 0;
  auto carve = [&](size_t bytes) -> char* {
    char* p = base + off;
    off = (off + bytes + 255) & ~(size_t)255;
    return p;
  };
  float* sW   = (float*)carve(512 * 4);
  float* sWp  = (float*)carve(512 * 4);
  float* fwn  = (float*)carve(512 * 4);
  float* h    = (float*)carve(64 * 1024 * 4);
  float* c    = (float*)carve(64 * 1024 * 4);
  float* gates= (float*)carve(64 * 4096 * 4);
  float* XWih = (float*)carve((size_t)896 * 4096 * 4);
  float* qy   = (float*)carve(64 * 512 * 4);
  float* qg   = (float*)carve(64 * 512 * 4);
  float* qpad = (float*)carve((size_t)896 * 320 * 4);   // K-padded q_feats
  unsigned short* wWy  = (unsigned short*)carve((size_t)512 * 2048 * 2);
  unsigned short* wWg  = (unsigned short*)carve((size_t)512 * 2048 * 2);
  unsigned short* wWhh = (unsigned short*)carve((size_t)4096 * 1024 * 2);
  unsigned short* wWih = (unsigned short*)carve((size_t)4096 * 320 * 2);

  // init state + output
  k_zero<<<(64 * 1024 + 255) / 256, 256, 0, stream>>>(h, 64 * 1024);
  k_zero<<<(64 * 1024 + 255) / 256, 256, 0, stream>>>(c, 64 * 1024);
  k_zero<<<1, 256, 0, stream>>>(counts, 64);

  // weight-norm scales
  k_rownorm<<<512, 256, 0, stream>>>(W_v,  W_g,  sW,  3072);
  k_rownorm<<<512, 256, 0, stream>>>(Wp_v, Wp_g, sWp, 3072);
  k_fwnorm<<<1, 256, 0, stream>>>(f_v, f_g, fwn);

  // bf16 B-fragment swizzles
  k_swizzle<<<(512 * 2048) / 256, 256, 0, stream>>>(W_v,  3072, 2048, sW,  wWy,  32, 2048);
  k_swizzle<<<(512 * 2048) / 256, 256, 0, stream>>>(Wp_v, 3072, 2048, sWp, wWg,  32, 2048);
  k_swizzle<<<(4096 * 1024) / 256, 256, 0, stream>>>(Whh, 1024, 1024, nullptr, wWhh, 256, 1024);
  k_swizzle<<<(4096 * 320) / 256, 256, 0, stream>>>(Wih,  300,  300, nullptr, wWih, 256, 320);

  // zero-pad q_feats K: [896,300] -> [896,320]
  {
    long total = (long)896 * 320;
    k_padcopy<<<(int)((total + 255) / 256), 256, 0, stream>>>(q_feats, 300, qpad,
                                                              320, total);
  }

  // hoisted input projection for all 14 timesteps: [896,320]x[320,4096]+bias
  k_gemm<<<dim3(56, 32), 256, 0, stream>>>(qpad, 320, 320, wWih, 256,
                                           bih, 0L, bhh, XWih, 4096);

  // sequential LSTM: gates = h@Whh^T + XWih[t] ; then cell update
  for (int t = 0; t < 14; ++t) {
    k_gemm<<<dim3(4, 32), 256, 0, stream>>>(h, 1024, 1024, wWhh, 256,
                                            XWih + (size_t)t * 4096, 14L * 4096,
                                            nullptr, gates, 4096);
    k_cell<<<(64 * 1024) / 256, 256, 0, stream>>>(gates, h, c);
  }

  // q-side projections (include biases)
  k_qproj<<<(2 * 64 * 512) / 256, 256, 0, stream>>>(h, W_v, Wp_v, sW, sWp,
                                                    W_b, Wp_b, qy, qg);

  // fused main pass: 32768 rows / 64 per WG
  k_fused<<<512, 256, 0, stream>>>(box_feats, wWy, wWg, qy, qg, fwn, indexv, counts);
}
